// P20RuntimeSequenceMixer_90048284328378
// MI455X (gfx1250) — matrix operand order repaired and checked by reading, not compiled, not run on hardware
//
#include <hip/hip_runtime.h>
#include <hip/hip_bf16.h>

// ---------------------------------------------------------------------------
// Problem constants
// ---------------------------------------------------------------------------
#define BB      4
#define SS      2048
#define DD      1024
#define BLOCKS  2
#define BW      512           // DD / BLOCKS
#define PAIRS   512           // DD / 2
#define PROJ_OUT 3584         // 3*DD + DD/2
#define MTOT    (BB * SS)     // 8192 GEMM rows

// Scan partitioning: 16 workgroups, 64 output columns each (one block half per WG)
#define NWG     16
#define COLS_PER_WG 64

// GEMM tiling: 64x64 workgroup tile, K-stage of 64, LDS rows padded 64->72 elems
#define KT      64
#define LDT     72

typedef __bf16 bf16_t;
typedef __attribute__((ext_vector_type(16))) __bf16 v16bf;
typedef __attribute__((ext_vector_type(8)))  __bf16 v8bf;
typedef __attribute__((ext_vector_type(4)))  __bf16 v4bf;
typedef __attribute__((ext_vector_type(8)))  float  v8f;
typedef __attribute__((ext_vector_type(4)))  float  v4f;

// TDM descriptor vector types (probe-documented builtin signature)
typedef __attribute__((ext_vector_type(4))) unsigned int tdm_u32x4;
typedef __attribute__((ext_vector_type(8))) int          tdm_i32x8;
typedef __attribute__((ext_vector_type(4))) int          tdm_i32x4;

#if defined(__has_builtin)
#  if __has_builtin(__builtin_amdgcn_tensor_load_to_lds)
#    define HAVE_TDM 1
#  else
#    define HAVE_TDM 0
#  endif
#else
#  define HAVE_TDM 0
#endif

__device__ __forceinline__ bf16_t f2bf(float f) {
    union { float f; unsigned u; } v; v.f = f;
    unsigned r = v.u + 0x7FFFu + ((v.u >> 16) & 1u);   // round-to-nearest-even
    union { unsigned short s; bf16_t b; } o; o.s = (unsigned short)(r >> 16);
    return o.b;
}

__device__ __forceinline__ v16bf cat8(v8bf a, v8bf b) {
    return __builtin_shufflevector(a, b, 0,1,2,3,4,5,6,7,8,9,10,11,12,13,14,15);
}

__device__ __forceinline__ float sigmoidf(float x) {
    return 1.0f / (1.0f + __expf(-x));
}

#if HAVE_TDM
// Issue one TDM 2D tile load: 64 rows x 64 bf16 elems, row-major source with
// element stride `ld`, into LDS with native padding (128B data + 16B pad per
// row => LDS row stride 144B == LDT elems). Call from ONE wave only (TDM
// ignores EXEC; duplicate issue would multiply the transfer).
__device__ __forceinline__ void tdm_load_tile64x64(unsigned lds_byte_addr,
                                                   const bf16_t* gsrc,
                                                   unsigned tensor_d0,
                                                   unsigned tensor_d1,
                                                   unsigned ld_elems) {
    unsigned long long ga = (unsigned long long)(uintptr_t)gsrc;
    tdm_u32x4 g0;
    g0[0] = 1u;                                   // count=1, user mode
    g0[1] = lds_byte_addr;                        // lds_addr
    g0[2] = (unsigned)(ga & 0xFFFFFFFFu);         // global_addr[31:0]
    g0[3] = (unsigned)((ga >> 32) & 0x1FFFFFFu)   // global_addr[56:32]
          | (2u << 30);                           // type = 2 ("image")
    tdm_i32x8 g1;
    g1[0] = (int)((1u << 16)      // data_size = 2 bytes (bf16)
                | (1u << 20)      // pad_enable
                | (4u << 22)      // pad_interval: 32 DWORDs (=128B row)
                | (3u << 25));    // pad_amount: 4 DWORDs (=16B)
    g1[1] = (int)((tensor_d0 & 0xFFFFu) << 16);                       // dim0[15:0]
    g1[2] = (int)(((tensor_d0 >> 16) & 0xFFFFu)
                | ((tensor_d1 & 0xFFFFu) << 16));                     // dim0 hi | dim1 lo
    g1[3] = (int)(((tensor_d1 >> 16) & 0xFFFFu) | (64u << 16));       // dim1 hi | tile_dim0=64
    g1[4] = (int)(64u);                                               // tile_dim1=64, tile_dim2=0
    g1[5] = (int)ld_elems;                                            // dim0_stride[31:0]
    g1[6] = 0;                                                        // stride hi / dim1_stride lo
    g1[7] = 0;
    tdm_i32x4 gz = {0, 0, 0, 0};
#if __clang_major__ >= 23
    tdm_i32x8 gz8 = {0, 0, 0, 0, 0, 0, 0, 0};
    __builtin_amdgcn_tensor_load_to_lds(g0, g1, gz, gz, gz8, 0);
#else
    __builtin_amdgcn_tensor_load_to_lds(g0, g1, gz, gz, 0);
#endif
}
#endif

// ---------------------------------------------------------------------------
// Kernel 0: vectorized f32 -> bf16 convert (x, proj_w are pre-converted once;
// bf16 inputs + f32 WMMA accumulate is the precision path the chip rewards)
// ---------------------------------------------------------------------------
__global__ void cvt_bf16_kernel(const float* __restrict__ in,
                                bf16_t* __restrict__ out, long long n4) {
    long long i = (long long)blockIdx.x * blockDim.x + threadIdx.x;
    const long long stride = (long long)gridDim.x * blockDim.x;
    for (; i < n4; i += stride) {
        v4f f = ((const v4f*)in)[i];
        v4bf b;
        #pragma unroll
        for (int j = 0; j < 4; ++j) b[j] = f2bf(f[j]);
        ((v4bf*)out)[i] = b;
    }
}

// ---------------------------------------------------------------------------
// Kernel 1: init scan state + grid-barrier counter (re-run every call ->
// deterministic across graph replays)
// ---------------------------------------------------------------------------
__global__ void init_state_kernel(const float* __restrict__ h_init,
                                  float* __restrict__ h0, int* __restrict__ counter) {
    int i = blockIdx.x * blockDim.x + threadIdx.x;
    if (i < BB * DD) h0[i] = h_init[i];
    if (i == 0) *counter = 0;
}

// ---------------------------------------------------------------------------
// Kernel 2: projection GEMM (bf16 WMMA, f32 accumulate) with fused activation
// epilogue. Workgroup tile 64x64, 8 waves (wave32), each wave owns 16x32.
// Tiles staged global->LDS by the Tensor Data Mover (wave 0 issues, all waves
// consume after s_wait_tensorcnt + barrier); b128 copy fallback otherwise.
// Gates written in scan layout [s][b][*].
// ---------------------------------------------------------------------------
__global__ __launch_bounds__(256)
void proj_gemm_kernel(const bf16_t* __restrict__ xbf,       // (B*S, D) bf16
                      const bf16_t* __restrict__ pw,        // (PROJ_OUT, D) bf16
                      const float* __restrict__ pb,         // (PROJ_OUT,) f32
                      float* __restrict__ ug,               // (S,B,D)
                      float* __restrict__ cg,               // (S,B,PAIRS)
                      float* __restrict__ sg,               // (S,B,PAIRS)
                      float* __restrict__ cd,               // (S,B,D)
                      float* __restrict__ og) {             // (S,B,D)
    __shared__ bf16_t lA[64][LDT];   // 64 M rows x 64 K (+8 pad)
    __shared__ bf16_t lB[64][LDT];   // 64 N rows x 64 K (+8 pad)

    const int tid  = threadIdx.x;
    const int wave = tid >> 5;
    const int lane = tid & 31;
    const int laneLo = lane & 15;
    const int laneHi = lane >> 4;
    const int wm = wave & 3;        // 4 M-tiles
    const int wn = wave >> 2;       // 2 N-groups of 32 cols
    const int mBase = blockIdx.x * 64;
    const int nBase = blockIdx.y * 64;

    v8f acc0 = {};
    v8f acc1 = {};

    for (int k0 = 0; k0 < DD; k0 += KT) {
#if HAVE_TDM
        if (tid < 32) {   // single wave issues both tile DMAs
            tdm_load_tile64x64((unsigned)(uintptr_t)&lA[0][0],
                               xbf + (size_t)mBase * DD + k0, DD, MTOT, DD);
            tdm_load_tile64x64((unsigned)(uintptr_t)&lB[0][0],
                               pw + (size_t)nBase * DD + k0, DD, PROJ_OUT, DD);
            __builtin_amdgcn_s_wait_tensorcnt(0);
        }
#else
        // vectorized fallback: 16B global loads -> 16B LDS stores
        for (int v = tid; v < 64 * (KT / 8); v += 256) {   // 512 vectors, 2/thread
            const int r = v >> 3, c = (v & 7) * 8;
            *(v8bf*)&lA[r][c] = *(const v8bf*)&xbf[(size_t)(mBase + r) * DD + k0 + c];
            *(v8bf*)&lB[r][c] = *(const v8bf*)&pw [(size_t)(nBase + r) * DD + k0 + c];
        }
#endif
        __syncthreads();

        #pragma unroll
        for (int kk = 0; kk < KT; kk += 32) {
            // A fragment: lane = M row; lo lanes K{0-7,16-23}, hi lanes K{8-15,24-31}
            const bf16_t* ra = &lA[wm * 16 + laneLo][kk + laneHi * 8];
            v16bf af = cat8(*(const v8bf*)ra, *(const v8bf*)(ra + 16));
            // B fragments: lane%16 = N col; lo lanes K{0-7,16-23}, hi lanes K{8-15,24-31}
            const bf16_t* rb0 = &lB[wn * 32 + laneLo][kk + laneHi * 8];
            v16bf bf0 = cat8(*(const v8bf*)rb0, *(const v8bf*)(rb0 + 16));
            const bf16_t* rb1 = &lB[wn * 32 + 16 + laneLo][kk + laneHi * 8];
            v16bf bf1 = cat8(*(const v8bf*)rb1, *(const v8bf*)(rb1 + 16));

            acc0 = __builtin_amdgcn_wmma_f32_16x16x32_bf16(false, af, false, bf0,
                                                           (short)0, acc0, false, false);
            acc1 = __builtin_amdgcn_wmma_f32_16x16x32_bf16(false, af, false, bf1,
                                                           (short)0, acc1, false, false);
        }
        __syncthreads();
    }

    // Epilogue: bias + activation, scatter into scan-layout gate arrays
    #pragma unroll
    for (int sub = 0; sub < 2; ++sub) {
        const v8f acc = sub ? acc1 : acc0;
        const int n = nBase + wn * 32 + sub * 16 + laneLo;
        const float bias = pb[n];
        #pragma unroll
        for (int r = 0; r < 8; ++r) {
            const int m = mBase + wm * 16 + r + laneHi * 8;   // = b*S + s
            const int b = m >> 11;          // /S
            const int s = m & (SS - 1);     // %S
            const float val = acc[r] + bias;
            const size_t sb = (size_t)s * BB + b;
            if (n < DD) {
                ug[sb * DD + n] = sigmoidf(val);
            } else if (n < DD + PAIRS) {
                const int p = n - DD;
                cg[sb * PAIRS + p] = __cosf(val);
                sg[sb * PAIRS + p] = __sinf(val);
            } else if (n < 2 * DD + PAIRS) {
                cd[sb * DD + (n - (DD + PAIRS))] = tanhf(val);
            } else {
                og[sb * DD + (n - (2 * DD + PAIRS))] = sigmoidf(val);
            }
        }
    }
}

// ---------------------------------------------------------------------------
// Kernel 3: persistent recurrent scan. 16 WGs x 128 threads (4 waves).
// Each WG keeps its 512x64 bf16 weight slice resident in LDS for all 2048
// steps; per step: rotate state (pairwise cos/sin) into a zero-padded 16x512
// bf16 A tile, 4 waves each chain 16 bf16 WMMAs (K=512) for one 16-col tile,
// gate-combine, write h_new (double-buffered) + emit, then grid barrier.
// ---------------------------------------------------------------------------
__global__ __launch_bounds__(128)
void scan_kernel(const bf16_t* __restrict__ wbf,     // (BLOCKS,BW,BW) bf16
                 const float* __restrict__ tbias,    // (D,)
                 const float* __restrict__ ug,
                 const float* __restrict__ cg,
                 const float* __restrict__ sg,
                 const float* __restrict__ cd,
                 const float* __restrict__ og,
                 float* __restrict__ h0buf,          // (B,D) ping
                 float* __restrict__ h1buf,          // (B,D) pong
                 int* __restrict__ counter,
                 float* __restrict__ out) {          // (B,S,D)
    __shared__ bf16_t lW[COLS_PER_WG][520];   // weight slice, K-major (+8 pad)
    __shared__ bf16_t lH[16][520];            // rotated state, rows 4..15 zero

    const int tid = threadIdx.x;
    const int wave = tid >> 5;
    const int lane = tid & 31;
    const int laneLo = lane & 15;
    const int laneHi = lane >> 4;
    const int colBase = blockIdx.x * COLS_PER_WG;   // global output column base
    const int g = colBase >> 9;                     // block index (0/1)
    const int oBase = colBase & (BW - 1);           // column base within block

    // one-time: weight slice resident in LDS (64 KB), 16B vector copies
    for (int v = tid; v < COLS_PER_WG * (BW / 8); v += 128) {
        const int o = v >> 6, c = (v & 63) * 8;
        *(v8bf*)&lW[o][c] =
            *(const v8bf*)&wbf[((size_t)g * BW + oBase + o) * BW + c];
    }
    // one-time: zero the whole A tile (incl. pads and M rows 4..15)
    for (int i = tid; i < 16 * 520; i += 128)
        (&lH[0][0])[i] = f2bf(0.0f);
    __syncthreads();

    for (int t = 0; t < SS; ++t) {
        const float* hin  = (t & 1) ? h1buf : h0buf;
        float*       hout = (t & 1) ? h0buf : h1buf;

        // rotate full block-half of state into bf16 A rows 0..3
        for (int i = tid; i < BB * BW; i += 128) {
            const int b = i >> 9, k = i & (BW - 1);
            const int d = (g << 9) + k;
            const int p = d >> 1;
            const size_t tb4 = (size_t)t * BB + b;
            const float c  = cg[tb4 * PAIRS + p];
            const float sn = sg[tb4 * PAIRS + p];
            const float ha = hin[b * DD + 2 * p];
            const float hb = hin[b * DD + 2 * p + 1];
            lH[b][k] = f2bf((d & 1) ? (ha * sn + hb * c) : (ha * c - hb * sn));
        }
        __syncthreads();

        // WMMA chain: wave owns 16 output cols, K = 512 in 16 steps
        v8f acc = {};
        const int n0 = wave * 16;
        #pragma unroll 4
        for (int k0 = 0; k0 < BW; k0 += 32) {
            const bf16_t* ra = &lH[laneLo][k0 + laneHi * 8];
            v16bf af = cat8(*(const v8bf*)ra, *(const v8bf*)(ra + 16));
            const bf16_t* rb = &lW[n0 + laneLo][k0 + laneHi * 8];
            v16bf bf_ = cat8(*(const v8bf*)rb, *(const v8bf*)(rb + 16));
            acc = __builtin_amdgcn_wmma_f32_16x16x32_bf16(false, af, false, bf_,
                                                          (short)0, acc, false, false);
        }

        // epilogue: gate combine + state update + emit
        const int d = colBase + n0 + laneLo;
        const float tb = tbias[d];
        #pragma unroll
        for (int r = 0; r < 8; ++r) {
            const int b = r + laneHi * 8;
            if (b < BB) {
                const float ht = acc[r] + tb;
                const size_t gidx = ((size_t)t * BB + b) * DD + d;
                const float u = ug[gidx];
                const float c = cd[gidx];
                const float o = og[gidx];
                const float hn = u * c + (1.0f - u) * ht;
                hout[b * DD + d] = hn;
                out[((size_t)b * SS + t) * DD + d] = o * hn;
            }
        }
        // hide next step's gate latency behind the barrier (L2-resident lines)
        if (t + 1 < SS) {
            const size_t nidx = ((size_t)(t + 1) * BB) * DD + d;
            __builtin_prefetch(&ug[nidx], 0, 0);
            __builtin_prefetch(&cd[nidx], 0, 0);
            __builtin_prefetch(&og[nidx], 0, 0);
            __builtin_prefetch(&cg[((size_t)(t + 1) * BB) * PAIRS + (d >> 1)], 0, 0);
        }
        __syncthreads();

        // device-wide barrier (monotonic counter, portable cluster-barrier stand-in)
        __threadfence();
        if (tid == 0) {
            atomicAdd(counter, 1);
            const int target = NWG * (t + 1);
            while (__hip_atomic_load(counter, __ATOMIC_ACQUIRE,
                                     __HIP_MEMORY_SCOPE_AGENT) < target) {
                __builtin_amdgcn_s_sleep(1);
            }
        }
        __syncthreads();
        __threadfence();
    }
}

// ---------------------------------------------------------------------------
// Host launcher
// ---------------------------------------------------------------------------
extern "C" void kernel_launch(void* const* d_in, const int* in_sizes, int n_in,
                              void* d_out, int out_size, void* d_ws, size_t ws_size,
                              hipStream_t stream) {
    const float* x       = (const float*)d_in[0];
    const float* proj_w  = (const float*)d_in[1];
    const float* proj_b  = (const float*)d_in[2];
    const float* tw      = (const float*)d_in[3];
    const float* tb      = (const float*)d_in[4];
    const float* h_init  = (const float*)d_in[5];
    float* out = (float*)d_out;

    // workspace carve-up
    char* ws = (char*)d_ws;
    size_t off = 0;
    auto carve = [&](size_t bytes) -> void* {
        void* p = ws + off;
        off += (bytes + 255) & ~(size_t)255;
        return p;
    };
    float*  ug    = (float*) carve((size_t)SS * BB * DD * 4);
    float*  cg    = (float*) carve((size_t)SS * BB * PAIRS * 4);
    float*  sg    = (float*) carve((size_t)SS * BB * PAIRS * 4);
    float*  cd    = (float*) carve((size_t)SS * BB * DD * 4);
    float*  og    = (float*) carve((size_t)SS * BB * DD * 4);
    bf16_t* xbf   = (bf16_t*)carve((size_t)MTOT * DD * 2);
    bf16_t* pwbf  = (bf16_t*)carve((size_t)PROJ_OUT * DD * 2);
    bf16_t* wbf   = (bf16_t*)carve((size_t)BLOCKS * BW * BW * 2);
    float*  h0    = (float*) carve((size_t)BB * DD * 4);
    float*  h1    = (float*) carve((size_t)BB * DD * 4);
    int*    cnt   = (int*)   carve(256);

    // 0: one-time bf16 conversions (x converted ONCE, not per N-tile)
    cvt_bf16_kernel<<<2048, 256, 0, stream>>>(x,      xbf,  (long long)MTOT * DD / 4);
    cvt_bf16_kernel<<<2048, 256, 0, stream>>>(proj_w, pwbf, (long long)PROJ_OUT * DD / 4);
    cvt_bf16_kernel<<<512,  256, 0, stream>>>(tw,     wbf,  (long long)BLOCKS * BW * BW / 4);

    // 1: state + barrier init
    init_state_kernel<<<(BB * DD + 255) / 256, 256, 0, stream>>>(h_init, h0, cnt);

    // 2: fused projection GEMM + activations (TDM-staged tiles)
    {
        dim3 grid(MTOT / 64, PROJ_OUT / 64);   // 128 x 56
        proj_gemm_kernel<<<grid, 256, 0, stream>>>(xbf, pwbf, proj_b,
                                                   ug, cg, sg, cd, og);
    }
    // 3: persistent recurrent scan (16 co-resident WGs)
    scan_kernel<<<NWG, 128, 0, stream>>>(wbf, tb, ug, cg, sg, cd, og,
                                         h0, h1, cnt, out);
}
